// NERModel_82772609728731
// MI455X (gfx1250) — matrix operand order repaired and checked by reading, not compiled, not run on hardware
//
#include <hip/hip_runtime.h>
#include <hip/hip_bf16.h>
#include <math.h>

// ---------------------------------------------------------------------------
// NER bi-LSTM on gfx1250 (MI455X): bf16 WMMA gate GEMMs, LDS-resident weights,
// gate-interleaved word weights + multi-WGP recurrence with global h exchange.
// ---------------------------------------------------------------------------

typedef __attribute__((ext_vector_type(16))) __bf16 v16bf;
typedef __attribute__((ext_vector_type(8)))  __bf16 v8bf;
typedef __attribute__((ext_vector_type(8)))  float  v8f;

#define BB     32
#define TT     128
#define WLEN   16
#define DW     300     // word emb dim
#define DC     100     // char emb dim
#define HC     100     // char LSTM hidden
#define HW     300     // word LSTM hidden
#define NTAGS  9
#define KC     200     // char gate K = DC + HC
#define KC_PAD 224     // padded to multiple of 32
#define NC     400     // 4*HC
#define XWID   500     // word LSTM input dim = DW + 2*HC
#define KW     800     // word gate K = XWID + HW (25*32, no pad)
#define NW     1200    // 4*HW
#define NW_PAD 1280    // gate-interleaved + padded to 80 n-tiles
#define NT_PER 10      // word n-tiles per block (80 / 8 n-slices)
#define DSLICE 40      // hidden dims per word block (NT_PER*16/4)
#define NSEQ   (BB*TT) // 4096

__device__ __forceinline__ float sigm_(float x) { return 1.0f / (1.0f + expf(-x)); }

// ---------------- weight prep: fp32 (K,N) -> bf16 transposed (N, Kpad) -----
__global__ void prep_wt(const float* __restrict__ src, __bf16* __restrict__ dst,
                        int K, int Kpad, int N) {
  int i = blockIdx.x * blockDim.x + threadIdx.x;
  if (i >= N * Kpad) return;
  int n = i / Kpad, k = i % Kpad;
  dst[i] = (k < K) ? (__bf16)src[(size_t)k * N + n] : (__bf16)0.0f;
}

// word weights: transposed + gate-interleaved columns n' = 4*d + g
// (so a 16-col n-tile holds complete i,j,f,o quadruples for 4 hidden dims)
__global__ void prep_wt_gate(const float* __restrict__ src, __bf16* __restrict__ dst,
                             int K, int H, int Npad) {
  int i = blockIdx.x * blockDim.x + threadIdx.x;
  if (i >= Npad * K) return;
  int c = i / K, k = i % K;
  int d = c >> 2, g = c & 3;
  dst[(size_t)c * K + k] =
      (d < H) ? (__bf16)src[(size_t)k * (4 * H) + g * H + d] : (__bf16)0.0f;
}

__global__ void zero_f32(float* __restrict__ p, int n) {
  int i = blockIdx.x * blockDim.x + threadIdx.x;
  if (i < n) p[i] = 0.0f;
}

// ---------------- fragment helpers (contiguous k = 16*(lane/16) + s, A and B alike) --------------------
__device__ __forceinline__ v16bf load_a_frag(const __bf16* arow, int kc, int lane) {
  int off8 = (lane >= 16) ? 16 : 0;
  const __bf16* p = arow + kc * 32 + off8;
  v8bf lo = *(const v8bf*)p;
  v8bf hi = *(const v8bf*)(p + 8);
  v16bf r;
#pragma unroll
  for (int i = 0; i < 8; ++i) { r[i] = lo[i]; r[i + 8] = hi[i]; }
  return r;
}
__device__ __forceinline__ v16bf load_b_frag(const __bf16* wcol, int kc) {
  const __bf16* q = wcol + kc * 32;
  v8bf lo = *(const v8bf*)q;
  v8bf hi = *(const v8bf*)(q + 8);
  v16bf r;
#pragma unroll
  for (int i = 0; i < 8; ++i) { r[i] = lo[i]; r[i + 8] = hi[i]; }
  return r;
}

// ---------------- char bi-LSTM -------------------------------------------
// grid = 256 blocks (bit0 = dir, rest = 32-sequence group), block = 64 (2 waves)
// Full bf16 weight matrix (400x224 = 179 KB) staged once into LDS.
__global__ void char_lstm(const int* __restrict__ char_ids,
                          const int* __restrict__ word_len,
                          const float* __restrict__ char_emb,
                          const __bf16* __restrict__ wt_fw,
                          const __bf16* __restrict__ wt_bw,
                          const float* __restrict__ b_fw,
                          const float* __restrict__ b_bw,
                          float* __restrict__ cf) {
  extern __shared__ __align__(16) char smem[];
  __bf16* Wl  = (__bf16*)smem;                 // 400 x 224   (179,200 B)
  __bf16* A   = Wl + NC * KC_PAD;              //  32 x 224   ( 14,336 B)
  float*  Z   = (float*)(A + 32 * KC_PAD);     //  32 x 400   ( 51,200 B)
  float*  Cst = Z + 32 * NC;                   //  32 x 100   ( 12,800 B)

  const int dir  = blockIdx.x & 1;
  const int seq0 = (blockIdx.x >> 1) * 32;
  const __bf16* wt   = dir ? wt_bw : wt_fw;
  const float*  bias = dir ? b_bw  : b_fw;

  const int tid  = threadIdx.x;
  const int lane = tid & 31;
  const int wave = tid >> 5;

  // one-time weight stage to LDS (16B chunks), init h/pad/c
  for (int i = tid; i < NC * KC_PAD / 8; i += blockDim.x)
    ((v8bf*)Wl)[i] = ((const v8bf*)wt)[i];
  for (int i = tid; i < 32 * (KC_PAD - DC); i += blockDim.x) {
    int r = i / (KC_PAD - DC), k = DC + i % (KC_PAD - DC);
    A[r * KC_PAD + k] = (__bf16)0.0f;
  }
  for (int i = tid; i < 32 * HC; i += blockDim.x) Cst[i] = 0.0f;
  __syncthreads();

  const __bf16* arow = A + (wave * 16 + (lane & 15)) * KC_PAD;
  const int colr = lane & 15;
  const int mo   = (lane >= 16) ? 8 : 0;
  const int koff = (lane >= 16) ? 16 : 0;

  for (int s = 0; s < WLEN; ++s) {
    // stage x_t (char embedding gather) into A cols [0, DC)
    for (int i = tid; i < 32 * DC; i += blockDim.x) {
      int r = i / DC, d = i % DC;
      int seq = seq0 + r;
      int L = word_len[seq];
      int pos = dir ? ((s < L) ? (L - 1 - s) : s) : s;
      int cid = char_ids[seq * WLEN + pos];
      A[r * KC_PAD + d] = (__bf16)char_emb[(size_t)cid * DC + d];
    }
    __syncthreads();

    // z = [x;h] @ W : 25 n-tiles x 7 k-chunks, A frags hoisted, B from LDS
    v16bf afr[7];
#pragma unroll
    for (int kc = 0; kc < 7; ++kc) afr[kc] = load_a_frag(arow, kc, lane);

    for (int nt = 0; nt < 25; ++nt) {
      v8f acc = {};
      const __bf16* wcol = Wl + (size_t)(nt * 16 + colr) * KC_PAD + koff;
#pragma unroll
      for (int kc = 0; kc < 7; ++kc) {
        v16bf bfr = load_b_frag(wcol, kc);
        acc = __builtin_amdgcn_wmma_f32_16x16x32_bf16(
            false, afr[kc], false, bfr, (short)0, acc, false, false);
      }
      int col = nt * 16 + colr;
#pragma unroll
      for (int r = 0; r < 8; ++r)
        Z[(wave * 16 + mo + r) * NC + col] = acc[r];
    }
    __syncthreads();

    // gate epilogue (i,j,f,o; forget_bias=1.0; state frozen for s >= len)
    for (int i = tid; i < 32 * HC; i += blockDim.x) {
      int r = i / HC, d = i % HC;
      int L = word_len[seq0 + r];
      float zi = Z[r * NC + d]          + bias[d];
      float zj = Z[r * NC + HC + d]     + bias[HC + d];
      float zf = Z[r * NC + 2 * HC + d] + bias[2 * HC + d];
      float zo = Z[r * NC + 3 * HC + d] + bias[3 * HC + d];
      float c  = Cst[r * HC + d];
      float cn = c * sigm_(zf + 1.0f) + sigm_(zi) * tanhf(zj);
      float hn = tanhf(cn) * sigm_(zo);
      if (s < L) {
        Cst[r * HC + d] = cn;
        A[r * KC_PAD + DC + d] = (__bf16)hn;
      }
    }
    __syncthreads();
  }

  for (int i = tid; i < 32 * HC; i += blockDim.x) {
    int r = i / HC, d = i % HC;
    cf[(size_t)(seq0 + r) * (2 * HC) + dir * HC + d] =
        (float)A[r * KC_PAD + DC + d];
  }
}

// ---------------- pack word-LSTM input x = [we | char_feat] as bf16 -------
__global__ void pack_x(const int* __restrict__ word_ids,
                       const float* __restrict__ word_emb,
                       const float* __restrict__ cf,
                       __bf16* __restrict__ xw) {
  int i = blockIdx.x * blockDim.x + threadIdx.x;
  if (i >= NSEQ * XWID) return;
  int bt = i / XWID, d = i % XWID;
  float v = (d < DW) ? word_emb[(size_t)word_ids[bt] * DW + d]
                     : cf[(size_t)bt * (2 * HC) + (d - DW)];
  xw[i] = (__bf16)v;
}

// ---------------- word bi-LSTM -------------------------------------------
// grid = 32 blocks: dir(2) x m-slice(2, 16 batches) x n-slice(8, 40 dims).
// Each block keeps its 160x800 bf16 weight slice (256 KB) in LDS. Gate
// interleave lets each block run its own epilogue. Recurrent h is exchanged
// through a double-buffered global array; the 8 blocks of each independent
// (dir, m-slice) group synchronize per step with an atomic arrive+spin
// barrier (blocks are 1/WGP at 294 KB LDS -> trivially co-resident).
__global__ void word_lstm(const int* __restrict__ seq_len,
                          const __bf16* __restrict__ xw,
                          const __bf16* __restrict__ wtg_fw,
                          const __bf16* __restrict__ wtg_bw,
                          const float* __restrict__ b_fw,
                          const float* __restrict__ b_bw,
                          float* __restrict__ hs_f,
                          float* __restrict__ hs_b,
                          float* __restrict__ hbuf,
                          unsigned* __restrict__ barcnt) {
  extern __shared__ __align__(16) char smem[];
  __bf16* Wl  = (__bf16*)smem;                 // 160 x 800  (256,000 B)
  __bf16* A   = Wl + NT_PER * 16 * KW;         //  16 x 800  ( 25,600 B)
  float*  Z   = (float*)(A + 16 * KW);         //  16 x 160  ( 10,240 B)
  float*  Cst = Z + 16 * NT_PER * 16;          //  16 x 40   (  2,560 B)

  const int blk = blockIdx.x;
  const int dir = blk >> 4;
  const int msl = (blk >> 3) & 1;
  const int nsl = blk & 7;
  const int grp = dir * 2 + msl;
  const __bf16* wt   = dir ? wtg_bw : wtg_fw;
  const float*  bias = dir ? b_bw  : b_fw;
  float*    hb_base = hbuf + (size_t)grp * 2 * 16 * 320;
  unsigned* cnt     = barcnt + grp;

  const int tid  = threadIdx.x;
  const int lane = tid & 31;
  const int wave = tid >> 5;

  // one-time weight-slice stage to LDS; c0 = 0
  const __bf16* wsrc = wt + (size_t)(nsl * NT_PER * 16) * KW;
  for (int i = tid; i < NT_PER * 16 * KW / 8; i += blockDim.x)
    ((v8bf*)Wl)[i] = ((const v8bf*)wsrc)[i];
  for (int i = tid; i < 16 * DSLICE; i += blockDim.x) Cst[i] = 0.0f;
  __syncthreads();

  const __bf16* arow = A + (lane & 15) * KW;
  const int colr = lane & 15;
  const int mo   = (lane >= 16) ? 8 : 0;
  const int koff = (lane >= 16) ? 16 : 0;

  for (int s = 0; s < TT; ++s) {
    const float* hrd = hb_base + (s & 1) * 16 * 320;        // h(s)
    float*       hwr = hb_base + ((s + 1) & 1) * 16 * 320;  // h(s+1)

    // stage x_t and h(s) into A = [x | h]
    for (int i = tid; i < 16 * XWID; i += blockDim.x) {
      int r = i / XWID, d = i % XWID;
      int b = msl * 16 + r;
      int L = seq_len[b];
      int t = dir ? ((s < L) ? (L - 1 - s) : s) : s;
      A[r * KW + d] = xw[((size_t)b * TT + t) * XWID + d];
    }
    for (int i = tid; i < 16 * HW; i += blockDim.x) {
      int r = i / HW, d = i % HW;
      A[r * KW + XWID + d] = (__bf16)hrd[r * 320 + d];
    }
    __syncthreads();

    // WMMA: wave owns n-tile nt = wave; dual accumulators for ILP
    {
      int nt = wave;
      v8f acc0 = {}, acc1 = {};
      const __bf16* wcol = Wl + (size_t)(nt * 16 + colr) * KW + koff;
#pragma unroll 5
      for (int kc = 0; kc < 25; kc += 2) {
        v16bf afr = load_a_frag(arow, kc, lane);
        v16bf bfr = load_b_frag(wcol, kc);
        acc0 = __builtin_amdgcn_wmma_f32_16x16x32_bf16(
            false, afr, false, bfr, (short)0, acc0, false, false);
        if (kc + 1 < 25) {
          v16bf afr1 = load_a_frag(arow, kc + 1, lane);
          v16bf bfr1 = load_b_frag(wcol, kc + 1);
          acc1 = __builtin_amdgcn_wmma_f32_16x16x32_bf16(
              false, afr1, false, bfr1, (short)0, acc1, false, false);
        }
      }
      int col = nt * 16 + colr;
#pragma unroll
      for (int r = 0; r < 8; ++r)
        Z[(mo + r) * (NT_PER * 16) + col] = acc0[r] + acc1[r];
    }
    __syncthreads();

    // gate epilogue on this block's 16 rows x 40 dims (gate-interleaved z)
    for (int i = tid; i < 16 * DSLICE; i += blockDim.x) {
      int r = i / DSLICE, dl = i % DSLICE;
      int d = nsl * DSLICE + dl;
      if (d < HW) {
        int b = msl * 16 + r;
        int L = seq_len[b];
        float zi = Z[r * (NT_PER * 16) + 4 * dl]     + bias[d];
        float zj = Z[r * (NT_PER * 16) + 4 * dl + 1] + bias[HW + d];
        float zf = Z[r * (NT_PER * 16) + 4 * dl + 2] + bias[2 * HW + d];
        float zo = Z[r * (NT_PER * 16) + 4 * dl + 3] + bias[3 * HW + d];
        float c  = Cst[r * DSLICE + dl];
        float cn = c * sigm_(zf + 1.0f) + sigm_(zi) * tanhf(zj);
        float hn = tanhf(cn) * sigm_(zo);
        bool  m  = (s < L);
        float hp = hrd[r * 320 + d];
        if (m) Cst[r * DSLICE + dl] = cn;
        hwr[r * 320 + d] = m ? hn : hp;               // frozen past length
        if (dir == 0) {
          hs_f[((size_t)b * TT + s) * HW + d] = m ? hn : 0.0f;
        } else if (m) {
          hs_b[((size_t)b * TT + (L - 1 - s)) * HW + d] = hn;
        }
      }
    }

    // per-group arrive + spin barrier (skip after final step)
    if (s != TT - 1) {
      __threadfence();
      __syncthreads();
      if (tid == 0) {
        __hip_atomic_fetch_add(cnt, 1u, __ATOMIC_RELEASE, __HIP_MEMORY_SCOPE_AGENT);
        unsigned target = 8u * (unsigned)(s + 1);
        while (__hip_atomic_load(cnt, __ATOMIC_ACQUIRE, __HIP_MEMORY_SCOPE_AGENT) < target)
          __builtin_amdgcn_s_sleep(2);
      }
      __syncthreads();
      __threadfence();
    }
  }
}

// ---------------- projection: logits = [of|ob] @ Wp + bp -------------------
__global__ void proj_k(const float* __restrict__ hs_f,
                       const float* __restrict__ hs_b,
                       const float* __restrict__ Wp,
                       const float* __restrict__ bp,
                       float* __restrict__ out) {
  int i = blockIdx.x * blockDim.x + threadIdx.x;
  if (i >= NSEQ * NTAGS) return;
  int bt = i / NTAGS, tg = i % NTAGS;
  float acc = bp[tg];
  const float* hf = hs_f + (size_t)bt * HW;
  const float* hb = hs_b + (size_t)bt * HW;
  for (int d = 0; d < HW; ++d) {
    acc += hf[d] * Wp[d * NTAGS + tg];
    acc += hb[d] * Wp[(d + HW) * NTAGS + tg];
  }
  out[i] = acc;
}

// ---------------------------------------------------------------------------
extern "C" void kernel_launch(void* const* d_in, const int* in_sizes, int n_in,
                              void* d_out, int out_size, void* d_ws, size_t ws_size,
                              hipStream_t stream) {
  (void)in_sizes; (void)n_in; (void)out_size; (void)ws_size;
  const int*   word_ids = (const int*)d_in[0];
  const int*   seq_len  = (const int*)d_in[1];
  const int*   char_ids = (const int*)d_in[2];
  const int*   word_len = (const int*)d_in[3];
  const float* word_emb = (const float*)d_in[4];
  const float* char_emb = (const float*)d_in[5];
  const float* Wc_fw    = (const float*)d_in[6];
  const float* bc_fw    = (const float*)d_in[7];
  const float* Wc_bw    = (const float*)d_in[8];
  const float* bc_bw    = (const float*)d_in[9];
  const float* Ww_fw    = (const float*)d_in[10];
  const float* bw_fw    = (const float*)d_in[11];
  const float* Ww_bw    = (const float*)d_in[12];
  const float* bw_bw    = (const float*)d_in[13];
  const float* W_proj   = (const float*)d_in[14];
  const float* b_proj   = (const float*)d_in[15];

  char* ws = (char*)d_ws;
  size_t off = 0;
  auto carve = [&](size_t bytes) -> void* {
    void* p = ws + off;
    off = (off + bytes + 255) & ~(size_t)255;
    return p;
  };
  __bf16*   wtc_fw = (__bf16*)carve((size_t)NC * KC_PAD * sizeof(__bf16));
  __bf16*   wtc_bw = (__bf16*)carve((size_t)NC * KC_PAD * sizeof(__bf16));
  __bf16*   wtw_fw = (__bf16*)carve((size_t)NW_PAD * KW * sizeof(__bf16));
  __bf16*   wtw_bw = (__bf16*)carve((size_t)NW_PAD * KW * sizeof(__bf16));
  __bf16*   xw     = (__bf16*)carve((size_t)NSEQ * XWID * sizeof(__bf16));
  float*    cf     = (float*) carve((size_t)NSEQ * 2 * HC * sizeof(float));
  float*    hs_f   = (float*) carve((size_t)NSEQ * HW * sizeof(float));
  float*    hs_b   = (float*) carve((size_t)NSEQ * HW * sizeof(float));
  float*    hbuf   = (float*) carve((size_t)4 * 2 * 16 * 320 * sizeof(float));
  unsigned* barcnt = (unsigned*)carve(4 * sizeof(unsigned));

  const int thr = 256;

  // 1) weight conversion (char: plain transpose; word: transpose + gate interleave)
  prep_wt<<<(NC * KC_PAD + thr - 1) / thr, thr, 0, stream>>>(Wc_fw, wtc_fw, KC, KC_PAD, NC);
  prep_wt<<<(NC * KC_PAD + thr - 1) / thr, thr, 0, stream>>>(Wc_bw, wtc_bw, KC, KC_PAD, NC);
  prep_wt_gate<<<(NW_PAD * KW + thr - 1) / thr, thr, 0, stream>>>(Ww_fw, wtw_fw, KW, HW, NW_PAD);
  prep_wt_gate<<<(NW_PAD * KW + thr - 1) / thr, thr, 0, stream>>>(Ww_bw, wtw_bw, KW, HW, NW_PAD);

  // 2) zero: hs_b tail positions, h exchange buffers, barrier counters
  zero_f32<<<(NSEQ * HW + thr - 1) / thr, thr, 0, stream>>>(hs_b, NSEQ * HW);
  zero_f32<<<(4 * 2 * 16 * 320 + thr - 1) / thr, thr, 0, stream>>>(hbuf, 4 * 2 * 16 * 320);
  zero_f32<<<1, thr, 0, stream>>>((float*)barcnt, 4);

  // 3) char bi-LSTM: 256 blocks x 64 threads, LDS-resident weights (~252 KB)
  size_t csh = (size_t)NC * KC_PAD * sizeof(__bf16) + 32 * KC_PAD * sizeof(__bf16)
             + 32 * NC * sizeof(float) + 32 * HC * sizeof(float);
  char_lstm<<<(NSEQ / 32) * 2, 64, csh, stream>>>(
      char_ids, word_len, char_emb, wtc_fw, wtc_bw, bc_fw, bc_bw, cf);

  // 4) pack x = [we | char_feat] to bf16
  pack_x<<<(NSEQ * XWID + thr - 1) / thr, thr, 0, stream>>>(word_ids, word_emb, cf, xw);

  // 5) word bi-LSTM: 32 blocks x 320 threads, LDS weight slices (~288 KB)
  size_t wsh = (size_t)NT_PER * 16 * KW * sizeof(__bf16) + 16 * KW * sizeof(__bf16)
             + 16 * NT_PER * 16 * sizeof(float) + 16 * DSLICE * sizeof(float);
  word_lstm<<<32, 320, wsh, stream>>>(seq_len, xw, wtw_fw, wtw_bw,
                                      bw_fw, bw_bw, hs_f, hs_b, hbuf, barcnt);

  // 6) projection
  proj_k<<<(NSEQ * NTAGS + thr - 1) / thr, thr, 0, stream>>>(
      hs_f, hs_b, W_proj, b_proj, (float*)d_out);
}
